// RIMCell_global_working_space_9835475108027
// MI455X (gfx1250) — matrix operand order repaired and checked by reading, not compiled, not run on hardware
//
#include <hip/hip_runtime.h>
#include <math.h>

typedef __attribute__((ext_vector_type(16))) _Float16 v16h;
typedef __attribute__((ext_vector_type(8)))  _Float16 v8h;
typedef __attribute__((ext_vector_type(8)))  float    v8f;

#define BATCH 1024
#define NSQ   64          // NS
#define SSQ   4           // S
#define DDQ   512         // D
#define NWQ   68          // NS + S
#define RSQRT32 0.17677669529663687f

enum { EPI_F32 = 0, EPI_F16 = 1, EPI_RELU_MEAN = 2, EPI_SIGMOID = 3, EPI_RES = 4 };

#define SHUF16(a, b) __builtin_shufflevector((a), (b), 0,1,2,3,4,5,6,7,8,9,10,11,12,13,14,15)

// ---------------------------------------------------------------------------
// Epilogue (shared by both GEMM variants)
// ---------------------------------------------------------------------------
template <int EPI>
__device__ __forceinline__
void gemm_epilogue(const v8f* acc, int NT_, int tn0base, int tm, int b, int l16, int half,
                   int M, int N, const float* __restrict__ bias,
                   float* __restrict__ outF, long strideF,
                   _Float16* __restrict__ outH, long strideH,
                   const float* __restrict__ res, long strideRes)
{
    for (int t = 0; t < NT_; ++t) {
        const int ncol = tn0base + t * 16 + l16;      // N always a multiple of 16
        const float bn = bias ? bias[ncol] : 0.0f;
#pragma unroll
        for (int r = 0; r < 8; ++r) {
            const int m = tm * 16 + half * 8 + r;
            if (m >= M) continue;
            float val = acc[t][r] + bn;
            if constexpr (EPI == EPI_RELU_MEAN) {
                val = fmaxf(val, 0.0f) * (1.0f / 64.0f);
                atomicAdd(outF + (long)b * N + ncol, val);
            } else if constexpr (EPI == EPI_F16) {
                outH[(long)b * strideH + (long)m * N + ncol] = (_Float16)val;
            } else {
                if constexpr (EPI == EPI_SIGMOID) val = 1.0f / (1.0f + __expf(-val));
                if constexpr (EPI == EPI_RES)     val += res[(long)b * strideRes + (long)m * N + ncol];
                outF[(long)b * strideF + (long)m * N + ncol] = val;
            }
        }
    }
}

// ---------------------------------------------------------------------------
// Register-direct WMMA GEMM (small N): one wave = 16 x (16*NT) tile.
// A-layout per lane: row = tm*16 + lane%16; halves at K = k0 + 8*(lane/16) (+16)
// B-layout per lane: n-col = lane%16; k-rows = k0 + 8*(lane/16) (+16), 8 each
// ---------------------------------------------------------------------------
template <int EPI, int NT>
__global__ __launch_bounds__(32)
void wmma_gemm(const _Float16* __restrict__ A, int lda, long strideA, int M,
               const _Float16* __restrict__ W16, int N, int K,
               const float* __restrict__ bias,
               float* __restrict__ outF, long strideF,
               _Float16* __restrict__ outH, long strideH,
               const float* __restrict__ res, long strideRes)
{
    const int lane = threadIdx.x;
    const int half = lane >> 4;
    const int l16  = lane & 15;
    const int tn0 = blockIdx.x * NT, tm = blockIdx.y, b = blockIdx.z;

    int arow = tm * 16 + l16;
    if (arow >= M) arow = M - 1;
    const _Float16* Ab = A + (long)b * strideA + (long)arow * lda;
    const _Float16* Bb = W16 + l16 + tn0 * 16;

    v8f acc[NT] = {};
    for (int k0 = 0; k0 < K; k0 += 32) {
        v8h a0 = *(const v8h*)(Ab + k0 + half * 8);
        v8h a1 = *(const v8h*)(Ab + k0 + 16 + half * 8);
        v16h av = SHUF16(a0, a1);
        const _Float16* Bk = Bb + (long)k0 * N;
#pragma unroll
        for (int t = 0; t < NT; ++t) {
            v16h bv;
#pragma unroll
            for (int s = 0; s < 16; ++s)
                bv[s] = Bk[(long)(half * 8 + (s & 7) + (s >> 3) * 16) * N + t * 16];
            acc[t] = __builtin_amdgcn_wmma_f32_16x16x32_f16(
                false, av, false, bv, (short)0, acc[t], false, false);
        }
    }
    gemm_epilogue<EPI>(acc, NT, tn0 * 16, tm, b, l16, half, M, N, bias,
                       outF, strideF, outH, strideH, res, strideRes);
}

// ---------------------------------------------------------------------------
// LDS-cooperative WMMA GEMM (N-span 512 per block): 8 waves x (16x64) tiles.
// The 16x128 A stage is loaded from global ONCE per block, double-buffered in
// LDS (row stride 136 halves to spread banks), broadcast to all 8 waves via
// ds_load_b128. Next stage's global load overlaps the current WMMA burst.
// Cuts A HBM traffic 8x on the L2-overflowing GEMMs (rh16: 268MB > 192MB L2).
// Requires K % 128 == 0 and N % 512 == 0.
// ---------------------------------------------------------------------------
template <int EPI>
__global__ __launch_bounds__(256)
void wmma_gemm_lds(const _Float16* __restrict__ A, int lda, long strideA, int M,
                   const _Float16* __restrict__ W16, int N, int K,
                   const float* __restrict__ bias,
                   float* __restrict__ outF, long strideF,
                   _Float16* __restrict__ outH, long strideH,
                   const float* __restrict__ res, long strideRes)
{
    __shared__ _Float16 smem[2][16 * 136];
    const int tid  = threadIdx.x;
    const int lane = tid & 31;
    const int wv   = tid >> 5;                // 0..7
    const int half = lane >> 4;
    const int l16  = lane & 15;
    const int tm = blockIdx.y, b = blockIdx.z;
    const int colbase = blockIdx.x * 512 + wv * 64;

    // staging role: each thread owns one 16B chunk of the 16x128 stage
    const int trow = tid >> 4;
    const int tcol = (tid & 15) * 8;
    int arow_t = tm * 16 + trow;
    if (arow_t >= M) arow_t = M - 1;
    const _Float16* Ag = A + (long)b * strideA + (long)arow_t * lda + tcol;

    const _Float16* Bw = W16 + l16 + colbase;

    v8f acc[4] = {};
    v8h stage = *(const v8h*)Ag;
    int buf = 0;
    for (int k0 = 0; k0 < K; k0 += 128) {
        *(v8h*)&smem[buf][trow * 136 + tcol] = stage;
        __syncthreads();
        if (k0 + 128 < K) stage = *(const v8h*)(Ag + k0 + 128);
#pragma unroll
        for (int kk = 0; kk < 4; ++kk) {
            v8h a0 = *(const v8h*)&smem[buf][l16 * 136 + kk * 32 + half * 8];
            v8h a1 = *(const v8h*)&smem[buf][l16 * 136 + kk * 32 + 16 + half * 8];
            v16h av = SHUF16(a0, a1);
            const _Float16* Bk = Bw + (long)(k0 + kk * 32) * N;
#pragma unroll
            for (int t = 0; t < 4; ++t) {
                v16h bv;
#pragma unroll
                for (int s = 0; s < 16; ++s)
                    bv[s] = Bk[(long)(half * 8 + (s & 7) + (s >> 3) * 16) * N + t * 16];
                acc[t] = __builtin_amdgcn_wmma_f32_16x16x32_f16(
                    false, av, false, bv, (short)0, acc[t], false, false);
            }
        }
        __syncthreads();
        buf ^= 1;
    }
    gemm_epilogue<EPI>(acc, 4, colbase, tm, b, l16, half, M, N, bias,
                       outF, strideF, outH, strideH, res, strideRes);
}

// ---------------------------------------------------------------------------
// Elementwise helpers
// ---------------------------------------------------------------------------
__global__ void f32_to_f16_kernel(const float* __restrict__ src,
                                  _Float16* __restrict__ dst, long n)
{
    long i = (long)blockIdx.x * blockDim.x + threadIdx.x;
    if (i < n) dst[i] = (_Float16)src[i];
}

__global__ void zero_f32_kernel(float* __restrict__ p, long n)
{
    long i = (long)blockIdx.x * blockDim.x + threadIdx.x;
    if (i < n) p[i] = 0.0f;
}

// wi = concat([h_s, mem], axis=1) in f16: (B, 68, 512)
__global__ void build_wi_kernel(const float* __restrict__ hs,
                                const float* __restrict__ mem,
                                _Float16* __restrict__ wi16)
{
    long i = (long)blockIdx.x * blockDim.x + threadIdx.x;
    const long total = (long)BATCH * NWQ * DDQ;
    if (i >= total) return;
    long b = i / ((long)NWQ * DDQ);
    int  r = (int)((i / DDQ) % NWQ);
    int  d = (int)(i % DDQ);
    float v = (r < NSQ) ? hs[((long)b * NSQ + r) * DDQ + d]
                        : mem[((long)b * SSQ + (r - NSQ)) * DDQ + d];
    wi16[i] = (_Float16)v;
}

// gin = inputs_mean (broadcast over s) + tanh(mem), f16 (B,4,512)
__global__ void gin_build_kernel(const float* __restrict__ umean,
                                 const float* __restrict__ mem,
                                 _Float16* __restrict__ gin16)
{
    long i = (long)blockIdx.x * blockDim.x + threadIdx.x;
    const long total = (long)BATCH * SSQ * DDQ;
    if (i >= total) return;
    long b = i / ((long)SSQ * DDQ);
    int  d = (int)(i % DDQ);
    gin16[i] = (_Float16)(umean[b * DDQ + d] + tanhf(mem[i]));
}

// mem_new = in_gate * tanh(mem_wo) + forget_gate * mem  -> f32 out + f16 copy
__global__ void mem_update_kernel(const float* __restrict__ gates,
                                  const float* __restrict__ memwo,
                                  const float* __restrict__ mem,
                                  float* __restrict__ out_mem,
                                  _Float16* __restrict__ memnew16)
{
    long i = (long)blockIdx.x * blockDim.x + threadIdx.x;
    const long total = (long)BATCH * SSQ * DDQ;
    if (i >= total) return;
    long bs = i / DDQ;
    int  d  = (int)(i % DDQ);
    const float* g = gates + bs * (2 * DDQ);
    float mn = g[d] * tanhf(memwo[i]) + g[DDQ + d] * mem[i];
    out_mem[i]   = mn;
    memnew16[i]  = (_Float16)mn;
}

// ---------------------------------------------------------------------------
// Window attention: per (b, s): softmax over n=68 of q.k/sqrt(32), then probs@v
// ---------------------------------------------------------------------------
__global__ __launch_bounds__(256)
void wattn_kernel(const float* __restrict__ q, const float* __restrict__ k,
                  const _Float16* __restrict__ v16, _Float16* __restrict__ memw16)
{
    const int b = blockIdx.x, s = blockIdx.y;
    const int tid = threadIdx.x;
    __shared__ float sc[NWQ];
    __shared__ float pr[NWQ];
    __shared__ float red[2];

    const float* qs = q + ((long)b * SSQ + s) * 32;
    if (tid < NWQ) {
        const float* kn = k + ((long)b * NWQ + tid) * 32;
        float d = 0.0f;
#pragma unroll
        for (int j = 0; j < 32; ++j) d += qs[j] * kn[j];
        sc[tid] = d * RSQRT32;
    }
    __syncthreads();
    if (tid == 0) {
        float m = sc[0];
        for (int n = 1; n < NWQ; ++n) m = fmaxf(m, sc[n]);
        float ssum = 0.0f;
        for (int n = 0; n < NWQ; ++n) ssum += __expf(sc[n] - m);
        red[0] = m; red[1] = ssum;
    }
    __syncthreads();
    if (tid < NWQ) pr[tid] = __expf(sc[tid] - red[0]) / red[1];
    __syncthreads();

    const _Float16* vb = v16 + (long)b * NWQ * DDQ;
    for (int d = tid; d < DDQ; d += 256) {
        float acc = 0.0f;
        for (int n = 0; n < NWQ; ++n) acc += pr[n] * (float)vb[(long)n * DDQ + d];
        memw16[((long)b * SSQ + s) * DDQ + d] = (_Float16)acc;
    }
}

// ---------------------------------------------------------------------------
// RIM attention: per (b, n): 4 heads x softmax over s=4, then probs@rv
// rh layout: (B, 64, 4*512) row-major, ready as A for the ro GEMM (K=2048)
// ---------------------------------------------------------------------------
__global__ __launch_bounds__(256)
void rattn_kernel(const float* __restrict__ rq, const float* __restrict__ rk,
                  const _Float16* __restrict__ rv16, _Float16* __restrict__ rh16)
{
    const int b = blockIdx.x, n = blockIdx.y;
    const int tid = threadIdx.x;
    __shared__ float pr[4][4];                 // [head][s]

    if (tid < 16) {
        int h = tid >> 2, s = tid & 3;
        const float* qp = rq + ((long)b * NSQ + n) * 128 + h * 32;
        const float* kp = rk + ((long)b * SSQ + s) * 128 + h * 32;
        float d = 0.0f;
#pragma unroll
        for (int j = 0; j < 32; ++j) d += qp[j] * kp[j];
        pr[h][s] = d * RSQRT32;
    }
    __syncthreads();
    if (tid < 4) {
        int h = tid;
        float m  = fmaxf(fmaxf(pr[h][0], pr[h][1]), fmaxf(pr[h][2], pr[h][3]));
        float e0 = __expf(pr[h][0] - m), e1 = __expf(pr[h][1] - m);
        float e2 = __expf(pr[h][2] - m), e3 = __expf(pr[h][3] - m);
        float inv = 1.0f / (e0 + e1 + e2 + e3);
        pr[h][0] = e0 * inv; pr[h][1] = e1 * inv; pr[h][2] = e2 * inv; pr[h][3] = e3 * inv;
    }
    __syncthreads();

    const _Float16* vb = rv16 + (long)b * SSQ * 2048;
    _Float16* out = rh16 + ((long)b * NSQ + n) * 2048;
    for (int hd = tid; hd < 2048; hd += 256) {
        int h = hd >> 9;
        float acc = pr[h][0] * (float)vb[0 * 2048 + hd]
                  + pr[h][1] * (float)vb[1 * 2048 + hd]
                  + pr[h][2] * (float)vb[2 * 2048 + hd]
                  + pr[h][3] * (float)vb[3 * 2048 + hd];
        out[hd] = (_Float16)acc;
    }
}

// ---------------------------------------------------------------------------
// Launcher
// ---------------------------------------------------------------------------
static inline long cdiv(long a, long b) { return (a + b - 1) / b; }

extern "C" void kernel_launch(void* const* d_in, const int* in_sizes, int n_in,
                              void* d_out, int out_size, void* d_ws, size_t ws_size,
                              hipStream_t stream)
{
    (void)in_sizes; (void)n_in; (void)out_size; (void)ws_size;

    const float* h_s  = (const float*)d_in[0];
    const float* mem  = (const float*)d_in[1];
    const float* wq_w = (const float*)d_in[2];  const float* wq_b = (const float*)d_in[3];
    const float* wk_w = (const float*)d_in[4];  const float* wk_b = (const float*)d_in[5];
    const float* wv_w = (const float*)d_in[6];  const float* wv_b = (const float*)d_in[7];
    const float* wo_w = (const float*)d_in[8];  const float* wo_b = (const float*)d_in[9];
    const float* ut_w = (const float*)d_in[10]; const float* ut_b = (const float*)d_in[11];
    const float* ug_w = (const float*)d_in[12]; const float* ug_b = (const float*)d_in[13];
    const float* rq_w = (const float*)d_in[14]; const float* rq_b = (const float*)d_in[15];
    const float* rk_w = (const float*)d_in[16]; const float* rk_b = (const float*)d_in[17];
    const float* rv_w = (const float*)d_in[18]; const float* rv_b = (const float*)d_in[19];
    const float* ro_w = (const float*)d_in[20]; const float* ro_b = (const float*)d_in[21];

    float* out_h   = (float*)d_out;                         // (B, 64, 512)
    float* out_mem = out_h + (long)BATCH * NSQ * DDQ;       // (B, 4, 512)

    // ---- workspace carve-up (256B aligned) ----
    char* ws = (char*)d_ws;
    size_t off = 0;
    auto alloc = [&](size_t bytes) -> void* {
        void* p = ws + off;
        off += (bytes + 255) & ~(size_t)255;
        return p;
    };
    _Float16* wq16 = (_Float16*)alloc((size_t)512 * 32 * 2);
    _Float16* wk16 = (_Float16*)alloc((size_t)512 * 32 * 2);
    _Float16* wv16 = (_Float16*)alloc((size_t)512 * 512 * 2);
    _Float16* wo16 = (_Float16*)alloc((size_t)512 * 512 * 2);
    _Float16* ut16 = (_Float16*)alloc((size_t)512 * 512 * 2);
    _Float16* ug16 = (_Float16*)alloc((size_t)512 * 1024 * 2);
    _Float16* rq16 = (_Float16*)alloc((size_t)512 * 128 * 2);
    _Float16* rk16 = (_Float16*)alloc((size_t)512 * 128 * 2);
    _Float16* rv16w = (_Float16*)alloc((size_t)512 * 2048 * 2);
    _Float16* ro16 = (_Float16*)alloc((size_t)2048 * 512 * 2);

    _Float16* wi16     = (_Float16*)alloc((size_t)BATCH * NWQ * DDQ * 2);
    float*    q_f      = (float*)   alloc((size_t)BATCH * SSQ * 32 * 4);
    float*    k_f      = (float*)   alloc((size_t)BATCH * NWQ * 32 * 4);
    _Float16* v16      = (_Float16*)alloc((size_t)BATCH * NWQ * DDQ * 2);
    _Float16* memw16   = (_Float16*)alloc((size_t)BATCH * SSQ * DDQ * 2);
    float*    memwo    = (float*)   alloc((size_t)BATCH * SSQ * DDQ * 4);
    float*    umean    = (float*)   alloc((size_t)BATCH * DDQ * 4);
    _Float16* gin16    = (_Float16*)alloc((size_t)BATCH * SSQ * DDQ * 2);
    float*    gates    = (float*)   alloc((size_t)BATCH * SSQ * 1024 * 4);
    _Float16* memnew16 = (_Float16*)alloc((size_t)BATCH * SSQ * DDQ * 2);
    float*    rq_f     = (float*)   alloc((size_t)BATCH * NSQ * 128 * 4);
    float*    rk_f     = (float*)   alloc((size_t)BATCH * SSQ * 128 * 4);
    _Float16* rv16a    = (_Float16*)alloc((size_t)BATCH * SSQ * 2048 * 2);
    _Float16* rh16     = (_Float16*)alloc((size_t)BATCH * NSQ * 2048 * 2);

    const dim3 W(32);
    const dim3 WG(256);
    const int MKV = BATCH * NWQ;   // 69632 = 16 * 4352 (k, v flattened)
    const int MS  = BATCH * SSQ;   // 4096  = 16 * 256  (wo, ug, rk, rv flattened)
    const int MH  = BATCH * NSQ;   // 65536 = 16 * 4096 (ro flattened)

    // ---- stage 0: precision conversion ----
#define CVT(src, dst, n) \
    f32_to_f16_kernel<<<dim3((unsigned)cdiv((n), 256)), 256, 0, stream>>>((src), (dst), (long)(n))
    CVT(wq_w, wq16, 512 * 32);
    CVT(wk_w, wk16, 512 * 32);
    CVT(wv_w, wv16, 512 * 512);
    CVT(wo_w, wo16, 512 * 512);
    CVT(ut_w, ut16, 512 * 512);
    CVT(ug_w, ug16, 512 * 1024);
    CVT(rq_w, rq16, 512 * 128);
    CVT(rk_w, rk16, 512 * 128);
    CVT(rv_w, rv16w, 512 * 2048);
    CVT(ro_w, ro16, 2048 * 512);
#undef CVT
    {
        long n = (long)BATCH * NWQ * DDQ;
        build_wi_kernel<<<dim3((unsigned)cdiv(n, 256)), 256, 0, stream>>>(h_s, mem, wi16);
        zero_f32_kernel<<<dim3((unsigned)cdiv((long)BATCH * DDQ, 256)), 256, 0, stream>>>(
            umean, (long)BATCH * DDQ);
    }

    // ---- window attention projections ----
    // q = mem @ wq (batched: mem rows are strided inside wi16)
    wmma_gemm<EPI_F32, 2><<<dim3(1, 1, BATCH), W, 0, stream>>>(
        wi16 + NSQ * DDQ, DDQ, (long)NWQ * DDQ, SSQ, wq16, 32, DDQ, wq_b,
        q_f, (long)SSQ * 32, nullptr, 0, nullptr, 0);
    // k = wi @ wk, flattened M = B*68 (exact 16-row tiles)
    wmma_gemm<EPI_F32, 2><<<dim3(1, MKV / 16, 1), W, 0, stream>>>(
        wi16, DDQ, 0, MKV, wk16, 32, DDQ, wk_b,
        k_f, 0, nullptr, 0, nullptr, 0);
    // v = wi @ wv -> f16, flattened, LDS-cooperative (A broadcast to 8 waves)
    wmma_gemm_lds<EPI_F16><<<dim3(1, MKV / 16, 1), WG, 0, stream>>>(
        wi16, DDQ, 0, MKV, wv16, DDQ, DDQ, wv_b,
        nullptr, 0, v16, 0, nullptr, 0);
    // inputs_mean = relu(h_s @ ut + b).mean(axis=1): batched (atomic epilogue needs b)
    wmma_gemm_lds<EPI_RELU_MEAN><<<dim3(1, 4, BATCH), WG, 0, stream>>>(
        wi16, DDQ, (long)NWQ * DDQ, NSQ, ut16, DDQ, DDQ, ut_b,
        umean, DDQ, nullptr, 0, nullptr, 0);

    // ---- attention + memory write path ----
    wattn_kernel<<<dim3(BATCH, SSQ), 256, 0, stream>>>(q_f, k_f, v16, memw16);
    // mem_w @ wo, flattened M = B*4
    wmma_gemm_lds<EPI_F32><<<dim3(1, MS / 16, 1), WG, 0, stream>>>(
        memw16, DDQ, 0, MS, wo16, DDQ, DDQ, wo_b,
        memwo, 0, nullptr, 0, nullptr, 0);
    {
        long n = (long)BATCH * SSQ * DDQ;
        gin_build_kernel<<<dim3((unsigned)cdiv(n, 256)), 256, 0, stream>>>(umean, mem, gin16);
    }
    // gates = sigmoid(gin @ ug + b), flattened
    wmma_gemm_lds<EPI_SIGMOID><<<dim3(2, MS / 16, 1), WG, 0, stream>>>(
        gin16, DDQ, 0, MS, ug16, 1024, DDQ, ug_b,
        gates, 0, nullptr, 0, nullptr, 0);
    {
        long n = (long)BATCH * SSQ * DDQ;
        mem_update_kernel<<<dim3((unsigned)cdiv(n, 256)), 256, 0, stream>>>(
            gates, memwo, mem, out_mem, memnew16);
    }

    // ---- RIM attention projections ----
    // rq = h_s @ rq_w: batched (h_s rows strided inside wi16), M=64 exact tiles
    wmma_gemm<EPI_F32, 4><<<dim3(2, 4, BATCH), W, 0, stream>>>(
        wi16, DDQ, (long)NWQ * DDQ, NSQ, rq16, 128, DDQ, rq_b,
        rq_f, (long)NSQ * 128, nullptr, 0, nullptr, 0);
    // rk = mem_new @ rk_w, flattened
    wmma_gemm<EPI_F32, 4><<<dim3(2, MS / 16, 1), W, 0, stream>>>(
        memnew16, DDQ, 0, MS, rk16, 128, DDQ, rk_b,
        rk_f, 0, nullptr, 0, nullptr, 0);
    // rv = mem_new @ rv_w -> f16, flattened
    wmma_gemm_lds<EPI_F16><<<dim3(4, MS / 16, 1), WG, 0, stream>>>(
        memnew16, DDQ, 0, MS, rv16w, 2048, DDQ, rv_b,
        nullptr, 0, rv16a, 0, nullptr, 0);

    rattn_kernel<<<dim3(BATCH, NSQ), 256, 0, stream>>>(rq_f, rk_f, rv16a, rh16);

    // ---- final projection + residual: h_new = h_s + rh @ ro + b (flattened) ----
    // A = rh16 (268MB, exceeds L2) is the traffic-critical operand: LDS variant
    // reads each A tile from HBM exactly once per block.
    wmma_gemm_lds<EPI_RES><<<dim3(1, MH / 16, 1), WG, 0, stream>>>(
        rh16, 2048, 0, MH, ro16, DDQ, 2048, ro_b,
        out_h, 0, nullptr, 0, h_s, 0);
}
